// MultiHeadAttention_17119739642336
// MI455X (gfx1250) — matrix operand order repaired and checked by reading, not compiled, not run on hardware
//
#include <hip/hip_runtime.h>
#include <hip/hip_bf16.h>
#include <stdint.h>

// ---------------------------------------------------------------------------
// Problem constants (reference: B,T,M,D = 2,2048,2048,128; H = 16)
// ---------------------------------------------------------------------------
constexpr int B_ = 2;
constexpr int T_ = 2048;
constexpr int M_ = 2048;
constexpr int D_ = 128;
constexpr int H_ = 16;
constexpr float QK_SCALE = 1.0f / 128.0f;
constexpr float NORM_EPS = 1e-6f;
constexpr float NEG_INF  = -1e30f;
constexpr float LOG_ROTARY_BASE = 9.21034037197618f; // ln(10000)

// ---------------------------------------------------------------------------
// WMMA vector types (CDNA5: v_wmma_f32_16x16x32_bf16)
// ---------------------------------------------------------------------------
typedef __attribute__((ext_vector_type(16))) __bf16 v16bf;
typedef __attribute__((ext_vector_type(8)))  __bf16 v8bf;
typedef __attribute__((ext_vector_type(8)))  float  v8f;

union ABf { v16bf v; v8bf h[2]; };

__device__ __forceinline__ __bf16 f2bf(float f) {
    union { float f; uint32_t u; } c; c.f = f;
    uint32_t u = c.u;
    uint32_t r = (u + 0x7FFFu + ((u >> 16) & 1u)) >> 16;   // RNE
    union { uint16_t q; __bf16 b; } o; o.q = (uint16_t)r;
    return o.b;
}

__device__ __forceinline__ v8f wmma_bf16(const ABf& a, const ABf& b, v8f c) {
    return __builtin_amdgcn_wmma_f32_16x16x32_bf16(
        false, a.v, false, b.v, (short)0, c, false, false);
}

// A operand (16x32 bf16): lane l -> row l%16; K chunks at (l<16?0:8), +16
__device__ __forceinline__ void load_a(ABf& a, const __bf16* rowptr,
                                       int k0, int half) {
    const __bf16* p = rowptr + k0 + (half ? 8 : 0);
    a.h[0] = *(const v8bf*)(p);
    a.h[1] = *(const v8bf*)(p + 16);
}
// B operand (32x16 bf16): lane l -> col l%16; K chunks at (l<16?0:8), +16
__device__ __forceinline__ void load_b(ABf& b, const __bf16* colptr,
                                       int k0, int half) {
    const __bf16* p = colptr + k0 + half * 8;
    b.h[0] = *(const v8bf*)(p);
    b.h[1] = *(const v8bf*)(p + 16);
}

// ---------------------------------------------------------------------------
// Kernel 1: fp32 -> bf16 elementwise convert (x matrix, keeps row-major)
// ---------------------------------------------------------------------------
__global__ void k_conv_bf16(const float* __restrict__ src,
                            __bf16* __restrict__ dst, size_t n) {
    size_t i = (size_t)blockIdx.x * blockDim.x + threadIdx.x;
    if (i < n) dst[i] = f2bf(src[i]);
}

// ---------------------------------------------------------------------------
// Kernel 2: fp32 [R,C] -> bf16 [C,R] transpose (weights -> K-contiguous B ops)
// ---------------------------------------------------------------------------
__global__ void k_transpose_bf16(const float* __restrict__ src,
                                 __bf16* __restrict__ dst, int R, int C) {
    __shared__ float tile[32][33];
    int x  = blockIdx.x * 32 + threadIdx.x;   // src col
    int y0 = blockIdx.y * 32;                 // src row base
    for (int i = threadIdx.y; i < 32; i += 8)
        tile[i][threadIdx.x] = src[(size_t)(y0 + i) * C + x];
    __syncthreads();
    int xo  = blockIdx.y * 32 + threadIdx.x;  // dst col (= src row)
    int yo0 = blockIdx.x * 32;                // dst row base (= src col)
    for (int i = threadIdx.y; i < 32; i += 8)
        dst[(size_t)(yo0 + i) * R + xo] = f2bf(tile[threadIdx.x][i]);
}

// ---------------------------------------------------------------------------
// Kernel 3: WMMA GEMM  C[MM, 2048] = A[MM, 2048] (bf16, row-major)
//                                  x Bt[2048, 2048] (bf16, K-contiguous cols)
// blockDim = 256 (8 waves, 4 along M x 2 along N); wave tile = 32x64;
// WG tile = 128x128. Double-buffered, branch-free K loop (last iteration
// peeled) so next-chunk loads overlap current WMMAs via partial
// s_wait_loadcnt. __launch_bounds__(256,1) lifts the occupancy register cap
// so the ~190 live VGPRs (64 acc + 96 operand buffers) never spill.
// ---------------------------------------------------------------------------
__global__ void __launch_bounds__(256, 1)
k_gemm_bf16(const __bf16* __restrict__ A,
            const __bf16* __restrict__ Bt,
            float* __restrict__ C) {
    constexpr int KK = M_;        // 2048
    constexpr int NN = M_;        // 2048
    const int lane  = threadIdx.x & 31;
    const int wave  = threadIdx.x >> 5;
    const int half  = lane >> 4;
    const int l16   = lane & 15;
    const int waveM = wave & 3;
    const int waveN = wave >> 2;
    const int mBase = blockIdx.x * 128 + waveM * 32;
    const int nBase = blockIdx.y * 128 + waveN * 64;

    const __bf16* arow[2];
    arow[0] = A + (size_t)(mBase + l16) * KK;
    arow[1] = A + (size_t)(mBase + 16 + l16) * KK;
    const __bf16* bcol[4];
#pragma unroll
    for (int nt = 0; nt < 4; ++nt)
        bcol[nt] = Bt + (size_t)(nBase + nt * 16 + l16) * KK;

    v8f acc[2][4] = {};
    ABf a0[2], b0[4], a1[2], b1[4];

    // prologue: buffer 0 <- k = 0
#pragma unroll
    for (int mt = 0; mt < 2; ++mt) load_a(a0[mt], arow[mt], 0, half);
#pragma unroll
    for (int nt = 0; nt < 4; ++nt) load_b(b0[nt], bcol[nt], 0, half);

    int k0 = 0;
    for (; k0 < KK - 64; k0 += 64) {
#pragma unroll
        for (int mt = 0; mt < 2; ++mt) load_a(a1[mt], arow[mt], k0 + 32, half);
#pragma unroll
        for (int nt = 0; nt < 4; ++nt) load_b(b1[nt], bcol[nt], k0 + 32, half);
#pragma unroll
        for (int mt = 0; mt < 2; ++mt)
#pragma unroll
            for (int nt = 0; nt < 4; ++nt)
                acc[mt][nt] = wmma_bf16(a0[mt], b0[nt], acc[mt][nt]);
#pragma unroll
        for (int mt = 0; mt < 2; ++mt) load_a(a0[mt], arow[mt], k0 + 64, half);
#pragma unroll
        for (int nt = 0; nt < 4; ++nt) load_b(b0[nt], bcol[nt], k0 + 64, half);
#pragma unroll
        for (int mt = 0; mt < 2; ++mt)
#pragma unroll
            for (int nt = 0; nt < 4; ++nt)
                acc[mt][nt] = wmma_bf16(a1[mt], b1[nt], acc[mt][nt]);
    }
    // peeled tail: k0 == KK - 64
#pragma unroll
    for (int mt = 0; mt < 2; ++mt) load_a(a1[mt], arow[mt], k0 + 32, half);
#pragma unroll
    for (int nt = 0; nt < 4; ++nt) load_b(b1[nt], bcol[nt], k0 + 32, half);
#pragma unroll
    for (int mt = 0; mt < 2; ++mt)
#pragma unroll
        for (int nt = 0; nt < 4; ++nt)
            acc[mt][nt] = wmma_bf16(a0[mt], b0[nt], acc[mt][nt]);
#pragma unroll
    for (int mt = 0; mt < 2; ++mt)
#pragma unroll
        for (int nt = 0; nt < 4; ++nt)
            acc[mt][nt] = wmma_bf16(a1[mt], b1[nt], acc[mt][nt]);

#pragma unroll
    for (int mt = 0; mt < 2; ++mt)
#pragma unroll
        for (int nt = 0; nt < 4; ++nt) {
            int col = nBase + nt * 16 + l16;
#pragma unroll
            for (int r = 0; r < 8; ++r)
                C[(size_t)(mBase + mt * 16 + r + 8 * half) * NN + col]
                    = acc[mt][nt][r];
        }
}

// ---------------------------------------------------------------------------
// Kernel 4: RMSNorm(q,k) + RoPE(q,k) + bf16 convert; v -> bf16 transposed.
// One wave per (b,t,h) row of D=128. Lane l owns dims {l, l+32, l+64, l+96}
// = rotary pairs (l, l+64) and (l+32, l+96). vT layout: [B,H,D,T].
// ---------------------------------------------------------------------------
__global__ void k_norm_rope(const float* __restrict__ qf,
                            const float* __restrict__ kf,
                            const float* __restrict__ vf,
                            __bf16* __restrict__ qb,
                            __bf16* __restrict__ kb,
                            __bf16* __restrict__ vT) {
    const int lane = threadIdx.x & 31;
    const int row  = blockIdx.x * (blockDim.x >> 5) + (threadIdx.x >> 5);
    if (row >= B_ * T_ * H_) return;
    const int h = row % H_;
    const int t = (row / H_) % T_;
    const int b = row / (H_ * T_);
    const size_t base = ((size_t)b * T_ + t) * M_ + h * D_;

    float c0, s0, c1, s1;
    __sincosf((float)t * __expf(-(float)lane        * (LOG_ROTARY_BASE / 64.f)), &s0, &c0);
    __sincosf((float)t * __expf(-(float)(lane + 32) * (LOG_ROTARY_BASE / 64.f)), &s1, &c1);

    const float* srcs[2] = { qf + base, kf + base };
    __bf16*      dsts[2] = { qb + base, kb + base };
#pragma unroll
    for (int s = 0; s < 2; ++s) {
        float e0 = srcs[s][lane];
        float e1 = srcs[s][lane + 32];
        float o0 = srcs[s][lane + 64];
        float o1 = srcs[s][lane + 96];
        float ss = e0 * e0 + e1 * e1 + o0 * o0 + o1 * o1;
        ss += __shfl_xor(ss, 1);  ss += __shfl_xor(ss, 2);
        ss += __shfl_xor(ss, 4);  ss += __shfl_xor(ss, 8);
        ss += __shfl_xor(ss, 16);
        float inv = rsqrtf(ss * (1.0f / 128.0f) + NORM_EPS);
        e0 *= inv; e1 *= inv; o0 *= inv; o1 *= inv;
        dsts[s][lane]      = f2bf(e0 * c0 - o0 * s0);
        dsts[s][lane + 32] = f2bf(e1 * c1 - o1 * s1);
        dsts[s][lane + 64] = f2bf(e0 * s0 + o0 * c0);
        dsts[s][lane + 96] = f2bf(e1 * s1 + o1 * c1);
    }
    const float* vrow = vf + base;
    __bf16* vcol = vT + ((size_t)(b * H_ + h) * D_) * T_;
#pragma unroll
    for (int d = 0; d < 4; ++d)
        vcol[(size_t)(lane + 32 * d) * T_ + t] = f2bf(vrow[lane + 32 * d]);
}

// ---------------------------------------------------------------------------
// Kernel 5: flash attention. blockDim = 128 (4 waves); wave = one 16-query
// tile. 32 keys/iter: 8 S-WMMAs + 8 PV-WMMAs. K-chunk B operands batched
// before their WMMAs (partial loadcnt waits); first 4 V operands issued
// before the softmax ALU + LDS P-relayout so their latency is hidden.
// __launch_bounds__(128,1) lifts the register cap (~200 live VGPRs peak).
// ---------------------------------------------------------------------------
__global__ void __launch_bounds__(128, 1)
k_attn(const __bf16* __restrict__ qb,
       const __bf16* __restrict__ kb,
       const __bf16* __restrict__ vT,
       __bf16* __restrict__ ob) {
    __shared__ __align__(16) __bf16 plds_all[4][16 * 40];
    const int lane = threadIdx.x & 31;
    const int wave = threadIdx.x >> 5;
    const int half = lane >> 4;
    const int l16  = lane & 15;
    const int h = blockIdx.y, b = blockIdx.z;
    const int q0 = (blockIdx.x * 4 + wave) * 16;
    __bf16* plds = plds_all[wave];

    // Q A-operands: 4 chunks covering K = D = 128
    ABf aQ[4];
    {
        const __bf16* qrow = qb + ((size_t)b * T_ + q0 + l16) * M_ + h * D_;
#pragma unroll
        for (int c = 0; c < 4; ++c) load_a(aQ[c], qrow, c * 32, half);
    }

    v8f acc[8] = {};
    float rmax[8], rsum[8];
#pragma unroll
    for (int r = 0; r < 8; ++r) { rmax[r] = NEG_INF; rsum[r] = 0.f; }

    const __bf16* vbase = vT + ((size_t)(b * H_ + h) * D_) * T_;

    for (int j0 = 0; j0 < q0 + 16; j0 += 32) {
        // ---- S = (q k^T) * scale, two 16-key tiles, causal-masked --------
        v8f s[2];
#pragma unroll
        for (int ts = 0; ts < 2; ++ts) {
            const int kvcol = j0 + ts * 16 + l16;   // this lane's key column
            const __bf16* krow = kb + ((size_t)b * T_ + kvcol) * M_ + h * D_;
            ABf bm[4];
#pragma unroll
            for (int c = 0; c < 4; ++c) load_b(bm[c], krow, c * 32, half);
            v8f sv = {};
#pragma unroll
            for (int c = 0; c < 4; ++c) sv = wmma_bf16(aQ[c], bm[c], sv);
#pragma unroll
            for (int r = 0; r < 8; ++r) {
                float val = sv[r] * QK_SCALE;
                int m = q0 + r + 8 * half;          // this vgpr's query row
                sv[r] = (kvcol > m) ? NEG_INF : val;
            }
            s[ts] = sv;
        }
        // ---- issue first half of the V operands early (independent of
        //      softmax); latency hides behind the ALU + LDS work below ----
        ABf bv[4];
#pragma unroll
        for (int nt = 0; nt < 4; ++nt)
            load_b(bv[nt], vbase + (size_t)(nt * 16 + l16) * T_, j0, half);

        // ---- online softmax (rows live in 16-lane halves) ----------------
        float scl[8];
#pragma unroll
        for (int r = 0; r < 8; ++r) {
            float tm = fmaxf(s[0][r], s[1][r]);
            tm = fmaxf(tm, __shfl_xor(tm, 1));
            tm = fmaxf(tm, __shfl_xor(tm, 2));
            tm = fmaxf(tm, __shfl_xor(tm, 4));
            tm = fmaxf(tm, __shfl_xor(tm, 8));
            float newm = fmaxf(rmax[r], tm);
            float sc = __expf(rmax[r] - newm);
            rmax[r] = newm;
            float p0 = __expf(s[0][r] - newm);
            float p1 = __expf(s[1][r] - newm);
            s[0][r] = p0; s[1][r] = p1;
            float psum = p0 + p1;
            psum += __shfl_xor(psum, 1);
            psum += __shfl_xor(psum, 2);
            psum += __shfl_xor(psum, 4);
            psum += __shfl_xor(psum, 8);
            rsum[r] = rsum[r] * sc + psum;
            scl[r] = sc;
        }
#pragma unroll
        for (int nt = 0; nt < 8; ++nt)
#pragma unroll
            for (int r = 0; r < 8; ++r) acc[nt][r] *= scl[r];

        // ---- P: C-layout f32 -> LDS -> A-layout bf16 ---------------------
#pragma unroll
        for (int ts = 0; ts < 2; ++ts)
#pragma unroll
            for (int r = 0; r < 8; ++r)
                plds[(r + 8 * half) * 40 + ts * 16 + l16] = f2bf(s[ts][r]);
        asm volatile("s_wait_dscnt 0" ::: "memory");
        ABf pa;
        {
            int kA = half ? 8 : 0;
            pa.h[0] = *(const v8bf*)(plds + l16 * 40 + kA);
            pa.h[1] = *(const v8bf*)(plds + l16 * 40 + kA + 16);
        }
        // ---- O += P * V (second half of V loads overlaps first WMMAs) ----
        ABf bv2[4];
#pragma unroll
        for (int nt = 0; nt < 4; ++nt)
            load_b(bv2[nt], vbase + (size_t)((nt + 4) * 16 + l16) * T_, j0, half);
#pragma unroll
        for (int nt = 0; nt < 4; ++nt)
            acc[nt] = wmma_bf16(pa, bv[nt], acc[nt]);
#pragma unroll
        for (int nt = 0; nt < 4; ++nt)
            acc[nt + 4] = wmma_bf16(pa, bv2[nt], acc[nt + 4]);
    }

    // ---- epilogue: normalize, store o as bf16 in [B,T,H*D] ---------------
    float rinv[8];
#pragma unroll
    for (int r = 0; r < 8; ++r) rinv[r] = 1.0f / rsum[r];
#pragma unroll
    for (int nt = 0; nt < 8; ++nt)
#pragma unroll
        for (int r = 0; r < 8; ++r)
            ob[((size_t)b * T_ + q0 + r + 8 * half) * M_ + h * D_ + nt * 16 + l16]
                = f2bf(acc[nt][r] * rinv[r]);
}

// ---------------------------------------------------------------------------
// Host launcher
// ---------------------------------------------------------------------------
extern "C" void kernel_launch(void* const* d_in, const int* in_sizes, int n_in,
                              void* d_out, int out_size, void* d_ws, size_t ws_size,
                              hipStream_t stream) {
    const float* x  = (const float*)d_in[0];   // [B,T,M]
    const float* wq = (const float*)d_in[1];   // [M,H*D]
    const float* wk = (const float*)d_in[2];
    const float* wv = (const float*)d_in[3];
    const float* wo = (const float*)d_in[4];   // [H*D,M]
    float* out = (float*)d_out;                // [B,T,M]

    constexpr size_t NX = (size_t)B_ * T_ * M_;       // 8,388,608
    constexpr size_t NW = (size_t)M_ * M_;            // 4,194,304
    char* ws = (char*)d_ws;
    size_t off = 0;
    auto alloc = [&](size_t bytes) {
        char* p = ws + off;
        off = (off + bytes + 255) & ~(size_t)255;
        return p;
    };
    __bf16* xb  = (__bf16*)alloc(NX * 2);
    __bf16* wqT = (__bf16*)alloc(NW * 2);
    __bf16* wkT = (__bf16*)alloc(NW * 2);
    __bf16* wvT = (__bf16*)alloc(NW * 2);
    __bf16* woT = (__bf16*)alloc(NW * 2);
    float*  qf  = (float*)alloc(NX * 4);
    float*  kf  = (float*)alloc(NX * 4);
    float*  vf  = (float*)alloc(NX * 4);
    __bf16* qb  = (__bf16*)alloc(NX * 2);
    __bf16* kbb = (__bf16*)alloc(NX * 2);
    __bf16* vT  = (__bf16*)alloc(NX * 2);
    __bf16* ob  = (__bf16*)alloc(NX * 2);

    // 1) x -> bf16
    k_conv_bf16<<<dim3((NX + 255) / 256), dim3(256), 0, stream>>>(x, xb, NX);

    // 2) weight transposes -> K-contiguous bf16 B operands
    dim3 tgrid(M_ / 32, M_ / 32), tblk(32, 8);
    k_transpose_bf16<<<tgrid, tblk, 0, stream>>>(wq, wqT, M_, M_);
    k_transpose_bf16<<<tgrid, tblk, 0, stream>>>(wk, wkT, M_, M_);
    k_transpose_bf16<<<tgrid, tblk, 0, stream>>>(wv, wvT, M_, M_);
    k_transpose_bf16<<<tgrid, tblk, 0, stream>>>(wo, woT, M_, M_);

    // 3) q,k,v projections: [B*T, M] x [M, H*D]
    const int MMrows = B_ * T_;
    dim3 ggrid(MMrows / 128, M_ / 128), gblk(256);
    k_gemm_bf16<<<ggrid, gblk, 0, stream>>>(xb, wqT, qf);
    k_gemm_bf16<<<ggrid, gblk, 0, stream>>>(xb, wkT, kf);
    k_gemm_bf16<<<ggrid, gblk, 0, stream>>>(xb, wvT, vf);

    // 4) RMSNorm + RoPE + convert (one wave per (b,t,h) row)
    const int nrows = B_ * T_ * H_;
    k_norm_rope<<<dim3(nrows / 8), dim3(256), 0, stream>>>(qf, kf, vf, qb, kbb, vT);

    // 5) flash attention
    k_attn<<<dim3(T_ / 64, H_, B_), dim3(128), 0, stream>>>(qb, kbb, vT, ob);

    // 6) output projection: [B*T, H*D] x [H*D, M] -> d_out
    k_gemm_bf16<<<ggrid, gblk, 0, stream>>>(ob, woT, out);
}
